// DirHGTConv_687194768176
// MI455X (gfx1250) — matrix operand order repaired and checked by reading, not compiled, not run on hardware
//
#include <hip/hip_runtime.h>
#include <math.h>

// ---------------------------------------------------------------------------
// DirHGTConv for gfx1250 (MI455X).  All dense GEMM work goes through
// v_wmma_f32_16x16x32_bf16 (wave32, f32 accumulate).  Operands are pre-packed
// into the exact CDNA5 WMMA VGPR fragment layout; the GEMM is register-blocked
// along N (NB tiles per wave) so each A fragment feeds NB wmma ops.
// ---------------------------------------------------------------------------

typedef __attribute__((ext_vector_type(16))) __bf16 v16bf;
typedef __attribute__((ext_vector_type(8)))  float  v8f;

#define HGT_F   256
#define HGT_H   8
#define HGT_D   32
#define HGT_KQV 768

// ---------------------------------------------------------------------------
// Pack a row-major fp32 matrix [N x Kc] (row stride `rowStride` floats) into
// bf16 WMMA *A-fragment* layout.  Per 16x32 tile (t, kb), lane L (0..31)
// stores 16 halves; half j maps to   m = L&15,
//   k = kb*32 + (j&1) + 2*((j>>1)&3) + ((j>>1)&4 ? 16 : 0) + 8*(L>>4)
// (ISA 05_wmma "16-bit A-Matrix 16x32").  Optional exact-gelu on load.
// ---------------------------------------------------------------------------
__global__ void pack_a_kernel(const float* __restrict__ src, long rowStride,
                              int N, int Kc, __bf16* __restrict__ dst,
                              int batch, long srcBatchStride, long dstBatchStride,
                              int gelu)
{
    long perB  = (long)N * Kc;
    long total = perB * batch;
    long tid   = (long)blockIdx.x * blockDim.x + threadIdx.x;
    if (tid >= total) return;
    int  b   = (int)(tid / perB);
    long idx = tid - (long)b * perB;
    int  j   = (int)(idx & 15);
    int  L   = (int)((idx >> 4) & 31);
    long rem = idx >> 9;
    int  KBt = Kc >> 5;
    int  kb  = (int)(rem % KBt);
    long t   = rem / KBt;
    int  m   = (int)(t * 16 + (L & 15));
    int  hi  = L >> 4;
    int  p   = j >> 1;
    int  k   = kb * 32 + (j & 1) + 2 * (p & 3) + ((p & 4) ? 16 : 0) + 8 * hi;
    float v = src[(long)b * srcBatchStride + (long)m * rowStride + k];
    if (gelu) v = 0.5f * v * (1.0f + erff(v * 0.70710678118654752f));
    dst[(long)b * dstBatchStride + idx] = (__bf16)v;
}

// ---------------------------------------------------------------------------
// Pack a row-major fp32 weight [K x Ncols] into bf16 WMMA *B-fragment*
// layout.  Per 32x16 tile (ct, kb), lane L stores column n = ct*16 + (L&15),
// half j holds K-row  k = kb*32 + 8*(L>>4) + (j < 8 ? j : 8 + j).
// ---------------------------------------------------------------------------
__global__ void pack_b_kernel(const float* __restrict__ src, int K, int Ncols,
                              __bf16* __restrict__ dst, int batch,
                              long srcBatchStride, long dstBatchStride)
{
    long perB  = (long)K * Ncols;
    long total = perB * batch;
    long tid   = (long)blockIdx.x * blockDim.x + threadIdx.x;
    if (tid >= total) return;
    int  b   = (int)(tid / perB);
    long idx = tid - (long)b * perB;
    int  j   = (int)(idx & 15);
    int  L   = (int)((idx >> 4) & 31);
    long rem = idx >> 9;
    int  KBt = K >> 5;
    int  kb  = (int)(rem % KBt);
    long ct  = rem / KBt;
    int  n   = (int)(ct * 16 + (L & 15));
    int  hi  = L >> 4;
    int  k   = kb * 32 + hi * 8 + (j < 8 ? j : 8 + j);
    dst[(long)b * dstBatchStride + idx] =
        (__bf16)src[(long)b * srcBatchStride + (long)k * Ncols + n];
}

// ---------------------------------------------------------------------------
// bf16 WMMA GEMM over pre-packed fragments, register-blocked along N:
// one wave computes a 16 x (16*NB) strip; per K-step it loads 1 A fragment +
// NB B fragments and issues NB v_wmma ops (A-reuse => 4x less L2 traffic
// for the streamed activation matrix at NB=4).
//  mode 0: out[m*outStride+col] = acc + bias[col]     (bias may be null)
//  mode 2: out-projection epilogue: bias + sigmoid-gated skip + fwd/bwd mix
// ---------------------------------------------------------------------------
template <int NB>
__global__ void gemm_wmma_bf16(const __bf16* __restrict__ pa,
                               const __bf16* __restrict__ pb,
                               float* __restrict__ out,
                               int Mtiles, int NtilesB, int KB, long outStride,
                               const float* __restrict__ bias, int mode,
                               const float* __restrict__ xres,
                               const float* __restrict__ skipP,
                               float mixScale, int accum)
{
    int waveId = blockIdx.x * (blockDim.x >> 5) + (threadIdx.x >> 5);
    if (waveId >= Mtiles * NtilesB) return;   // whole-wave exit: EXEC stays all-1s
    int mt    = waveId / NtilesB;
    int nbIdx = waveId % NtilesB;
    int lane  = threadIdx.x & 31;

    const v16bf* aP = (const v16bf*)(pa + (((long)mt * KB) * 32 + lane) * 16);
    const v16bf* bP[NB];
#pragma unroll
    for (int j = 0; j < NB; ++j)
        bP[j] = (const v16bf*)(pb + (((long)(nbIdx * NB + j) * KB) * 32 + lane) * 16);

    v8f acc[NB] = {};
    for (int kb = 0; kb < KB; ++kb) {
        v16bf a = aP[(long)kb * 32];          // 512 halves per K-step
#pragma unroll
        for (int j = 0; j < NB; ++j) {
            v16bf b = bP[j][(long)kb * 32];
            acc[j] = __builtin_amdgcn_wmma_f32_16x16x32_bf16(false, a, false, b,
                                                             (short)0, acc[j],
                                                             false, false);
        }
    }

    int n  = lane & 15;
    int hi = lane >> 4;

    if (mode == 2) {
        float aG = 1.0f / (1.0f + expf(-skipP[0]));
#pragma unroll
        for (int j = 0; j < NB; ++j) {
            int   col  = (nbIdx * NB + j) * 16 + n;
            float bval = bias ? bias[col] : 0.0f;
#pragma unroll
            for (int r = 0; r < 8; ++r) {
                int   m   = mt * 16 + r + 8 * hi;     // C/D layout: row = r + 8*hi
                float o   = acc[j][r] + bval;
                float res = aG * o + (1.0f - aG) * xres[(long)m * HGT_F + col];
                long  oi  = (long)m * outStride + col;
                if (accum) out[oi] += mixScale * res;
                else       out[oi]  = mixScale * res;
            }
        }
    } else {
#pragma unroll
        for (int j = 0; j < NB; ++j) {
            int   col  = (nbIdx * NB + j) * 16 + n;
            float bval = bias ? bias[col] : 0.0f;
#pragma unroll
            for (int r = 0; r < 8; ++r) {
                int m = mt * 16 + r + 8 * hi;
                out[(long)m * outStride + col] = acc[j][r] + bval;
            }
        }
    }
}

// ---------------------------------------------------------------------------
// Utility / edge kernels
// ---------------------------------------------------------------------------
__global__ void fill_f32(float* __restrict__ p, long n, float v)
{
    long i = (long)blockIdx.x * blockDim.x + threadIdx.x;
    if (i < n) p[i] = v;
}

__global__ void fixmax_kernel(float* __restrict__ p, long n)
{
    long i = (long)blockIdx.x * blockDim.x + threadIdx.x;
    if (i < n) { float v = p[i]; if (!isfinite(v)) p[i] = 0.0f; }
}

__device__ __forceinline__ void atomicMaxFloat(float* addr, float val)
{
    // sign-split trick: IEEE ordering via int max / uint min
    if (val >= 0.0f) atomicMax((int*)addr, __float_as_int(val));
    else             atomicMin((unsigned int*)addr, __float_as_uint(val));
}

__global__ void edge_logits_kernel(const int* __restrict__ srcI, const int* __restrict__ dstI,
                                   int E, const float* __restrict__ kqv, int dstOff,
                                   const float* __restrict__ kt,
                                   const float* __restrict__ prel,
                                   float* __restrict__ ex, float* __restrict__ segmax)
{
    int tid = blockIdx.x * blockDim.x + threadIdx.x;
    if (tid >= E * HGT_H) return;
    int e = tid >> 3, h = tid & 7;
    int s = srcI[e], d = dstI[e];
    const float4* q4 = (const float4*)(kqv + ((long)(dstOff + d)) * HGT_KQV + HGT_F + h * HGT_D);
    const float4* k4 = (const float4*)(kt + (long)s * HGT_F + h * HGT_D);
    float acc = 0.0f;
#pragma unroll
    for (int i = 0; i < 8; ++i) {
        float4 a = q4[i], b = k4[i];
        acc += a.x * b.x + a.y * b.y + a.z * b.z + a.w * b.w;
    }
    float logit = acc * prel[h] * 0.17677669529663687f;   // 1/sqrt(32)
    ex[(long)e * HGT_H + h] = logit;
    atomicMaxFloat(segmax + ((long)(dstOff + d)) * HGT_H + h, logit);
}

__global__ void edge_expsum_kernel(const int* __restrict__ dstI, int E, int dstOff,
                                   float* __restrict__ ex,
                                   const float* __restrict__ segmax,
                                   float* __restrict__ denom)
{
    int tid = blockIdx.x * blockDim.x + threadIdx.x;
    if (tid >= E * HGT_H) return;
    int e = tid >> 3, h = tid & 7;
    int d = dstI[e];
    long gi = ((long)(dstOff + d)) * HGT_H + h;
    float v = expf(ex[(long)e * HGT_H + h] - segmax[gi]);
    ex[(long)e * HGT_H + h] = v;
    atomicAdd(denom + gi, v);
}

__global__ void edge_agg_kernel(const int* __restrict__ srcI, const int* __restrict__ dstI,
                                int E, int dstOff, const float* __restrict__ ex,
                                const float* __restrict__ denom,
                                const float* __restrict__ vt, float* __restrict__ agg)
{
    int tid = blockIdx.x * blockDim.x + threadIdx.x;
    if (tid >= E * HGT_H) return;
    int e = tid >> 3, h = tid & 7;
    int s = srcI[e], d = dstI[e];
    long gi = ((long)(dstOff + d)) * HGT_H + h;
    float alpha = ex[(long)e * HGT_H + h] / (denom[gi] + 1e-16f);
    const float4* v4 = (const float4*)(vt + (long)s * HGT_F + h * HGT_D);
    float* o = agg + ((long)(dstOff + d)) * HGT_F + h * HGT_D;
#pragma unroll
    for (int i = 0; i < 8; ++i) {
        float4 v = v4[i];
        atomicAdd(o + 4 * i + 0, alpha * v.x);
        atomicAdd(o + 4 * i + 1, alpha * v.y);
        atomicAdd(o + 4 * i + 2, alpha * v.z);
        atomicAdd(o + 4 * i + 3, alpha * v.w);
    }
}

// ---------------------------------------------------------------------------
// Host orchestration
// ---------------------------------------------------------------------------
static inline long cdiv(long a, long b) { return (a + b - 1) / b; }

extern "C" void kernel_launch(void* const* d_in, const int* in_sizes, int n_in,
                              void* d_out, int out_size, void* d_ws, size_t ws_size,
                              hipStream_t stream)
{
    (void)n_in; (void)out_size; (void)ws_size;

    const float* x_data = (const float*)d_in[0];
    const float* x_inst = (const float*)d_in[1];
    const int*   ei[4]  = {(const int*)d_in[2], (const int*)d_in[3],
                           (const int*)d_in[4], (const int*)d_in[5]};
    int Ecnt[4] = {in_sizes[2] / 2, in_sizes[3] / 2, in_sizes[4] / 2, in_sizes[5] / 2};

    const int ND = in_sizes[0] / HGT_F;
    const int NI = in_sizes[1] / HGT_F;
    const int NT = ND + NI;
    const int Nnt[2]  = {ND, NI};
    const long gOff[2] = {0, (long)ND};
    const float* xptr[2] = {x_data, x_inst};
    long Etot = (long)Ecnt[0] + Ecnt[1] + Ecnt[2] + Ecnt[3];

    // params_{fwd,bwd} leaves, jax sorted-pytree order (22 leaves each):
    //  0-3 k_rel (et sorted), 4-7 kqv{data,inst}{W,b}, 8-11 out_lin{data,inst}{W,b},
    //  12-15 p_rel, 16-17 skip{data,inst}, 18-21 v_rel
    auto P = [&](int pass, int leaf) { return (const float*)d_in[6 + pass * 22 + leaf]; };

    // ---- workspace layout -------------------------------------------------
    char*  ws  = (char*)d_ws;
    size_t off = 0;
    auto alloc = [&](size_t bytes) { size_t o = off; off += (bytes + 255) & ~(size_t)255; return o; };

    __bf16* XB[2];   XB[0]   = (__bf16*)(ws + alloc((size_t)ND * HGT_F * 2));
                     XB[1]   = (__bf16*)(ws + alloc((size_t)NI * HGT_F * 2));
    __bf16* WBkqv[2];WBkqv[0]= (__bf16*)(ws + alloc((size_t)HGT_F * HGT_KQV * 2));
                     WBkqv[1]= (__bf16*)(ws + alloc((size_t)HGT_F * HGT_KQV * 2));
    __bf16* WBout[2];WBout[0]= (__bf16*)(ws + alloc((size_t)HGT_F * HGT_F * 2));
                     WBout[1]= (__bf16*)(ws + alloc((size_t)HGT_F * HGT_F * 2));
    __bf16* KRP     = (__bf16*)(ws + alloc((size_t)4 * 8 * 1024 * 2));
    __bf16* VRP     = (__bf16*)(ws + alloc((size_t)4 * 8 * 1024 * 2));
    float*  kqvF    = (float*) (ws + alloc((size_t)NT * HGT_KQV * 4));
    __bf16* KP[2];   KP[0]   = (__bf16*)(ws + alloc((size_t)ND * HGT_F * 2));
                     KP[1]   = (__bf16*)(ws + alloc((size_t)NI * HGT_F * 2));
    __bf16* VP[2];   VP[0]   = (__bf16*)(ws + alloc((size_t)ND * HGT_F * 2));
                     VP[1]   = (__bf16*)(ws + alloc((size_t)NI * HGT_F * 2));
    int NmaxSrc = ND > NI ? ND : NI;
    float*  ktF     = (float*) (ws + alloc((size_t)NmaxSrc * HGT_F * 4));
    float*  vtF     = (float*) (ws + alloc((size_t)NmaxSrc * HGT_F * 4));
    float*  exF     = (float*) (ws + alloc((size_t)Etot * HGT_H * 4));
    float*  segmaxF = (float*) (ws + alloc((size_t)NT * HGT_H * 4));
    float*  denomF  = (float*) (ws + alloc((size_t)NT * HGT_H * 4));
    // agg + packed-gelu aliased into kqvF region (kqv dead after logits stage)
    float*  aggF    = kqvF;
    __bf16* GP[2];
    GP[0] = (__bf16*)((char*)kqvF + (size_t)NT * HGT_F * 4);
    GP[1] = (__bf16*)((char*)kqvF + (size_t)NT * HGT_F * 4 + (size_t)ND * HGT_F * 2);

    // ---- one-time: pack x into A-fragments --------------------------------
    for (int nt = 0; nt < 2; ++nt) {
        long tot = (long)Nnt[nt] * HGT_F;
        pack_a_kernel<<<(int)cdiv(tot, 256), 256, 0, stream>>>(
            xptr[nt], HGT_F, Nnt[nt], HGT_F, XB[nt], 1, 0, 0, 0);
    }

    // NB-dispatching GEMM launcher (NB=4 for the 768/256-wide GEMMs,
    // NB=2 for the 32-wide relation GEMMs)
    auto gemm = [&](const __bf16* pa, const __bf16* pb, float* out,
                    int Mt, int Ntl, int KB, long ostride, const float* bias,
                    int mode, const float* xres, const float* skipP,
                    float mix, int accum) {
        if ((Ntl & 3) == 0) {
            int waves = Mt * (Ntl >> 2);
            gemm_wmma_bf16<4><<<(waves + 7) / 8, 256, 0, stream>>>(
                pa, pb, out, Mt, Ntl >> 2, KB, ostride, bias, mode, xres, skipP, mix, accum);
        } else if ((Ntl & 1) == 0) {
            int waves = Mt * (Ntl >> 1);
            gemm_wmma_bf16<2><<<(waves + 7) / 8, 256, 0, stream>>>(
                pa, pb, out, Mt, Ntl >> 1, KB, ostride, bias, mode, xres, skipP, mix, accum);
        } else {
            int waves = Mt * Ntl;
            gemm_wmma_bf16<1><<<(waves + 7) / 8, 256, 0, stream>>>(
                pa, pb, out, Mt, Ntl, KB, ostride, bias, mode, xres, skipP, mix, accum);
        }
    };

    struct EM { const int* s; const int* d; int E; int snt; int dnt; int pi; long eoff; };

    for (int pass = 0; pass < 2; ++pass) {
        const float mixScale = 0.5f;   // (1-ALPHA) and ALPHA are both 0.5
        const int   accum    = pass;

        EM em[4];
        if (pass == 0) {  // fwd: control(i->i), input(d->i), output(i->d), call(i->i)
            em[0] = {ei[0],           ei[0] + Ecnt[0], Ecnt[0], 1, 1, 2, 0};
            em[1] = {ei[1],           ei[1] + Ecnt[1], Ecnt[1], 0, 1, 0, 0};
            em[2] = {ei[2],           ei[2] + Ecnt[2], Ecnt[2], 1, 0, 3, 0};
            em[3] = {ei[3],           ei[3] + Ecnt[3], Ecnt[3], 1, 1, 1, 0};
        } else {          // bwd: reversed edges
            em[0] = {ei[0] + Ecnt[0], ei[0],           Ecnt[0], 1, 1, 2, 0};
            em[1] = {ei[1] + Ecnt[1], ei[1],           Ecnt[1], 1, 0, 3, 0};
            em[2] = {ei[2] + Ecnt[2], ei[2],           Ecnt[2], 0, 1, 0, 0};
            em[3] = {ei[3] + Ecnt[3], ei[3],           Ecnt[3], 1, 1, 1, 0};
        }
        { long eo = 0; for (int i = 0; i < 4; ++i) { em[i].eoff = eo; eo += em[i].E; } }

        // ---- pack this pass's weights ------------------------------------
        pack_b_kernel<<<(int)cdiv((long)HGT_F * HGT_KQV, 256), 256, 0, stream>>>(
            P(pass, 4), HGT_F, HGT_KQV, WBkqv[0], 1, 0, 0);
        pack_b_kernel<<<(int)cdiv((long)HGT_F * HGT_KQV, 256), 256, 0, stream>>>(
            P(pass, 6), HGT_F, HGT_KQV, WBkqv[1], 1, 0, 0);
        pack_b_kernel<<<(int)cdiv((long)HGT_F * HGT_F, 256), 256, 0, stream>>>(
            P(pass, 8), HGT_F, HGT_F, WBout[0], 1, 0, 0);
        pack_b_kernel<<<(int)cdiv((long)HGT_F * HGT_F, 256), 256, 0, stream>>>(
            P(pass, 10), HGT_F, HGT_F, WBout[1], 1, 0, 0);
        for (int i = 0; i < 4; ++i) {   // per-head 32x32 relation matrices
            pack_b_kernel<<<(int)cdiv(8L * 1024, 256), 256, 0, stream>>>(
                P(pass, 0 + em[i].pi), 32, 32, KRP + (long)i * 8192, 8, 1024, 1024);
            pack_b_kernel<<<(int)cdiv(8L * 1024, 256), 256, 0, stream>>>(
                P(pass, 18 + em[i].pi), 32, 32, VRP + (long)i * 8192, 8, 1024, 1024);
        }

        // ---- KQV projection (WMMA) ---------------------------------------
        for (int nt = 0; nt < 2; ++nt)
            gemm(XB[nt], WBkqv[nt], kqvF + gOff[nt] * HGT_KQV,
                 Nnt[nt] / 16, HGT_KQV / 16, HGT_F / 32, HGT_KQV,
                 P(pass, nt ? 7 : 5), 0, nullptr, nullptr, 0.f, 0);

        // ---- pack K / V slices (per-head) for relation GEMMs -------------
        for (int nt = 0; nt < 2; ++nt) {
            long tot = (long)Nnt[nt] * HGT_D * HGT_H;
            pack_a_kernel<<<(int)cdiv(tot, 256), 256, 0, stream>>>(
                kqvF + gOff[nt] * HGT_KQV, HGT_KQV, Nnt[nt], HGT_D,
                KP[nt], HGT_H, HGT_D, (long)Nnt[nt] * HGT_D, 0);
            pack_a_kernel<<<(int)cdiv(tot, 256), 256, 0, stream>>>(
                kqvF + gOff[nt] * HGT_KQV + 2 * HGT_F, HGT_KQV, Nnt[nt], HGT_D,
                VP[nt], HGT_H, HGT_D, (long)Nnt[nt] * HGT_D, 0);
        }

        // ---- segment softmax: max ----------------------------------------
        fill_f32<<<(int)cdiv((long)NT * HGT_H, 256), 256, 0, stream>>>(segmaxF, (long)NT * HGT_H, -INFINITY);
        fill_f32<<<(int)cdiv((long)NT * HGT_H, 256), 256, 0, stream>>>(denomF,  (long)NT * HGT_H, 0.0f);

        for (int i = 0; i < 4; ++i) {
            int snt = em[i].snt, Nsrc = Nnt[snt];
            for (int h = 0; h < HGT_H; ++h)   // k_t = K @ k_rel[h]  (WMMA)
                gemm(KP[snt] + (long)h * Nsrc * HGT_D, KRP + (long)i * 8192 + h * 1024,
                     ktF + h * HGT_D, Nsrc / 16, 2, 1, HGT_F,
                     nullptr, 0, nullptr, nullptr, 0.f, 0);
            int dstOff = em[i].dnt ? ND : 0;
            edge_logits_kernel<<<(int)cdiv((long)em[i].E * HGT_H, 256), 256, 0, stream>>>(
                em[i].s, em[i].d, em[i].E, kqvF, dstOff, ktF,
                P(pass, 12 + em[i].pi), exF + em[i].eoff * HGT_H, segmaxF);
        }
        fixmax_kernel<<<(int)cdiv((long)NT * HGT_H, 256), 256, 0, stream>>>(segmaxF, (long)NT * HGT_H);

        // ---- exp & denom --------------------------------------------------
        for (int i = 0; i < 4; ++i) {
            int dstOff = em[i].dnt ? ND : 0;
            edge_expsum_kernel<<<(int)cdiv((long)em[i].E * HGT_H, 256), 256, 0, stream>>>(
                em[i].d, em[i].E, dstOff, exF + em[i].eoff * HGT_H, segmaxF, denomF);
        }

        // ---- aggregate (kqvF now dead -> agg aliases it) -----------------
        fill_f32<<<(int)cdiv((long)NT * HGT_F, 256), 256, 0, stream>>>(aggF, (long)NT * HGT_F, 0.0f);
        for (int i = 0; i < 4; ++i) {
            int snt = em[i].snt, Nsrc = Nnt[snt];
            for (int h = 0; h < HGT_H; ++h)   // v_t = V @ v_rel[h]  (WMMA)
                gemm(VP[snt] + (long)h * Nsrc * HGT_D, VRP + (long)i * 8192 + h * 1024,
                     vtF + h * HGT_D, Nsrc / 16, 2, 1, HGT_F,
                     nullptr, 0, nullptr, nullptr, 0.f, 0);
            int dstOff = em[i].dnt ? ND : 0;
            edge_agg_kernel<<<(int)cdiv((long)em[i].E * HGT_H, 256), 256, 0, stream>>>(
                em[i].s, em[i].d, em[i].E, dstOff, exF + em[i].eoff * HGT_H,
                denomF, vtF, aggF);
        }

        // ---- gelu + pack, out-projection (WMMA) + skip + mix -------------
        for (int nt = 0; nt < 2; ++nt) {
            long tot = (long)Nnt[nt] * HGT_F;
            pack_a_kernel<<<(int)cdiv(tot, 256), 256, 0, stream>>>(
                aggF + gOff[nt] * HGT_F, HGT_F, Nnt[nt], HGT_F, GP[nt], 1, 0, 0, 1);
            gemm(GP[nt], WBout[nt], (float*)d_out + gOff[nt] * HGT_F,
                 Nnt[nt] / 16, HGT_F / 16, HGT_F / 32, HGT_F,
                 P(pass, nt ? 11 : 9), 2, xptr[nt], P(pass, nt ? 17 : 16),
                 mixScale, accum);
        }
    }
}
